// MoELayer_6468220748458
// MI455X (gfx1250) — compile-verified, matrix operand order checked by reading, not run on hardware
//
#include <hip/hip_runtime.h>
#include <hip/hip_bf16.h>

// ---------------------------------------------------------------------------
// MoE layer for MI455X (gfx1250): sparse top-2 routing + bf16 WMMA expert FFN.
// 32-token tiles, H chunked through LDS, 4-chain WMMA groups with A/B reuse:
// 1 global b128 + 1 ds b128 per v_wmma_f32_16x16x32_bf16.
// ---------------------------------------------------------------------------

typedef __attribute__((ext_vector_type(16))) __bf16 v16bf;
typedef __attribute__((ext_vector_type(8)))  __bf16 v8bf;
typedef __attribute__((ext_vector_type(8)))  float  v8f;

#define T_TOK   8192            // B*S
#define DIM     1024            // D
#define HID     4096            // H
#define NEXP    8               // experts
#define TILE_M  32              // token tile = 2 WMMA M-subtiles
#define HCH     1024            // H chunk held in LDS
#define MAXTILE (T_TOK / TILE_M)   // 256 tiles per expert worst case

// ---- workspace layout (bytes) ----
#define OFF_CNT   ((size_t)0)                                   // 8 x int
#define OFF_FSUM  ((size_t)64)                                  // 8 x float
#define OFF_PSUM  ((size_t)128)                                 // 8 x float
#define OFF_TOK   ((size_t)256)                                 // NEXP*T ints
#define OFF_WGT   (OFF_TOK  + (size_t)NEXP * T_TOK * 4)         // NEXP*T floats
#define OFF_XBF   (OFF_WGT  + (size_t)NEXP * T_TOK * 4)         // T*D bf16
#define OFF_W1BF  (OFF_XBF  + (size_t)T_TOK * DIM * 2)          // E*H*D bf16 (transposed)
#define OFF_W2BF  (OFF_W1BF + (size_t)NEXP * DIM * HID * 2)     // E*D*H bf16 (transposed)

// ---- LDS layout (bytes) ----
#define LDS_XS    ((size_t)0)                                   // 32 x 1024 bf16 = 64 KB
#define LDS_HS    (LDS_XS + (size_t)TILE_M * DIM * 2)           // 32 x 1024 bf16 = 64 KB
#define LDS_YS    (LDS_HS + (size_t)TILE_M * HCH * 2)           // 128 units x 32 x 8 f32 = 128 KB
#define LDS_TKS   (LDS_YS + (size_t)128 * 32 * 8 * 4)
#define LDS_WTS   (LDS_TKS + TILE_M * 4)
#define LDS_TOTAL (LDS_WTS + TILE_M * 4)                        // 262400 B

// ---------------------------------------------------------------------------
// fp32 -> bf16 bulk convert (grid-stride), for x
// ---------------------------------------------------------------------------
__global__ void f32_to_bf16(const float* __restrict__ src,
                            __bf16* __restrict__ dst, size_t n) {
  size_t i      = (size_t)blockIdx.x * blockDim.x + threadIdx.x;
  size_t stride = (size_t)gridDim.x * blockDim.x;
  for (; i < n; i += stride) dst[i] = (__bf16)src[i];
}

// ---------------------------------------------------------------------------
// fp32 [R][C] -> bf16 [C][R] tiled transpose (coalesced both sides),
// one expert slice per blockIdx.z.
// ---------------------------------------------------------------------------
#define TT 32
__global__ __launch_bounds__(256)
void convert_transpose(const float* __restrict__ src, __bf16* __restrict__ dst,
                       int R, int C) {
  __shared__ __bf16 tile[TT][TT + 2];   // +2 bf16 pad: 17-bank row shift
  const float* s = src + (size_t)blockIdx.z * R * C;
  __bf16*      d = dst + (size_t)blockIdx.z * R * C;
  const int c0 = blockIdx.x * TT;
  const int r0 = blockIdx.y * TT;
  const int tc = threadIdx.x & 31;   // 0..31
  const int tr = threadIdx.x >> 5;   // 0..7
#pragma unroll
  for (int i = 0; i < TT; i += 8)
    tile[tr + i][tc] = (__bf16)s[(size_t)(r0 + tr + i) * C + (c0 + tc)];
  __syncthreads();
#pragma unroll
  for (int i = 0; i < TT; i += 8)
    d[(size_t)(c0 + tr + i) * R + (r0 + tc)] = tile[tc][tr + i];
}

// ---------------------------------------------------------------------------
// Router: logits -> softmax -> top-2 -> normalized weights; builds per-expert
// token lists + accumulators for the load-balancing aux loss.
// ---------------------------------------------------------------------------
__global__ void router_kernel(const float* __restrict__ x,
                              const float* __restrict__ Wg,
                              int* __restrict__ cnt,
                              float* __restrict__ fsum,
                              float* __restrict__ psum,
                              int* __restrict__ tokList,
                              float* __restrict__ wList) {
  int t = blockIdx.x * blockDim.x + threadIdx.x;
  if (t >= T_TOK) return;

  float logit[NEXP];
#pragma unroll
  for (int e = 0; e < NEXP; ++e) logit[e] = 0.0f;

  const float* xt = x + (size_t)t * DIM;
  for (int d = 0; d < DIM; ++d) {
    float xv = xt[d];
    const float* wr = Wg + (size_t)d * NEXP;
#pragma unroll
    for (int e = 0; e < NEXP; ++e) logit[e] += xv * wr[e];
  }

  float mx = logit[0];
#pragma unroll
  for (int e = 1; e < NEXP; ++e) mx = fmaxf(mx, logit[e]);
  float p[NEXP], s = 0.0f;
#pragma unroll
  for (int e = 0; e < NEXP; ++e) { p[e] = __expf(logit[e] - mx); s += p[e]; }
  float inv = 1.0f / s;
#pragma unroll
  for (int e = 0; e < NEXP; ++e) p[e] *= inv;

  int i0 = 0;
#pragma unroll
  for (int e = 1; e < NEXP; ++e) if (p[e] > p[i0]) i0 = e;
  int i1 = (i0 == 0) ? 1 : 0;
#pragma unroll
  for (int e = 0; e < NEXP; ++e) if (e != i0 && p[e] > p[i1]) i1 = e;

  float w0 = p[i0], w1 = p[i1];
  float wn = 1.0f / (w0 + w1);
  w0 *= wn; w1 *= wn;

#pragma unroll
  for (int e = 0; e < NEXP; ++e) atomicAdd(&psum[e], p[e]);
  atomicAdd(&fsum[i0], 1.0f);
  atomicAdd(&fsum[i1], 1.0f);

  int p0 = atomicAdd(&cnt[i0], 1);
  tokList[i0 * T_TOK + p0] = t;  wList[i0 * T_TOK + p0] = w0;
  int p1 = atomicAdd(&cnt[i1], 1);
  tokList[i1 * T_TOK + p1] = t;  wList[i1 * T_TOK + p1] = w1;
}

__global__ void finalize_aux(const float* __restrict__ fsum,
                             const float* __restrict__ psum,
                             float* __restrict__ out_aux) {
  float s = 0.0f;
#pragma unroll
  for (int e = 0; e < NEXP; ++e)
    s += (fsum[e] / (float)T_TOK) * (psum[e] / (float)T_TOK);
  *out_aux = (float)NEXP * s;
}

// ---------------------------------------------------------------------------
// Fragment load: bf16 K-octets p[0..7] and p[16..23] (two 16B vector loads, no packing ops).
// ---------------------------------------------------------------------------
__device__ __forceinline__ v16bf ldfrag(const __bf16* p) {
  v8bf lo = *(const v8bf*)p;
  v8bf hi = *(const v8bf*)(p + 16);
  return __builtin_shufflevector(lo, hi, 0, 1, 2, 3, 4, 5, 6, 7,
                                         8, 9, 10, 11, 12, 13, 14, 15);
}

__device__ __forceinline__ v8f wmma_bf16(v16bf a, v16bf b, v8f c) {
  return __builtin_amdgcn_wmma_f32_16x16x32_bf16(
      false, a, false, b, (short)0, c, false, false);
}

__device__ __forceinline__ float gelu_erf(float v) {
  return 0.5f * v * (1.0f + erff(v * 0.70710678118654752f));
}

// ---------------------------------------------------------------------------
// Expert FFN: one block = (expert e, tile of 32 tokens routed to e).
// For each 1024-wide H chunk:
//   GEMM1: Hs[32,1024] = gelu(X[32,1024] @ W1t_chunk + b1)     (LDS)
//   GEMM2: Ys[32,1024] += Hs @ W2t_chunk                       (LDS, frag layout)
// Last chunk fuses b2 + gating-weighted atomic scatter to out.
// Inner groups: 4 WMMA chains share 2 A frags + 2 B frags.
// ---------------------------------------------------------------------------
__global__ __launch_bounds__(256)
void expert_ffn(const __bf16* __restrict__ xbf,
                const __bf16* __restrict__ W1t,   // [E][H][D]
                const __bf16* __restrict__ W2t,   // [E][D][H]
                const float* __restrict__ b1,
                const float* __restrict__ b2,
                const int* __restrict__ cnt,
                const int* __restrict__ tokList,
                const float* __restrict__ wList,
                float* __restrict__ out) {
  extern __shared__ char smem[];
  __bf16* Xs  = (__bf16*)(smem + LDS_XS);
  __bf16* Hs  = (__bf16*)(smem + LDS_HS);
  float*  Ys  = (float*) (smem + LDS_YS);
  int*    tks = (int*)   (smem + LDS_TKS);
  float*  wts = (float*) (smem + LDS_WTS);

  const int e    = blockIdx.x / MAXTILE;
  const int tile = blockIdx.x % MAXTILE;
  const int count = cnt[e];
  if (tile * TILE_M >= count) return;   // block-uniform exit

  const int tid = threadIdx.x;
  if (tid < TILE_M) {
    int idx = tile * TILE_M + tid;
    int cl  = (idx < count) ? idx : (count - 1);
    tks[tid] = tokList[e * T_TOK + cl];
    wts[tid] = (idx < count) ? wList[e * T_TOK + cl] : 0.0f;  // pad rows: weight 0
  }
  __syncthreads();

  // stage gathered X tile into LDS (16-byte chunks; rows are 2KB contiguous)
  for (int i = tid; i < TILE_M * (DIM / 8); i += blockDim.x) {
    int r  = i / (DIM / 8);
    int c8 = i % (DIM / 8);
    *(uint4*)&Xs[r * DIM + c8 * 8] =
        *(const uint4*)&xbf[(size_t)tks[r] * DIM + (size_t)c8 * 8];
  }

  const int lane = tid & 31;
  const int wave = tid >> 5;
  const int r16  = lane & 15;          // A: M row / B,C,D: N column within tile
  const int kh   = (lane >> 4) << 3;   // 0 or 8: K-octet owned by this half-wave

  const __bf16* W1e = W1t + (size_t)e * DIM * HID;
  const __bf16* W2e = W2t + (size_t)e * HID * DIM;

  for (int hb = 0; hb < HID; hb += HCH) {
    __syncthreads();   // Xs ready (1st iter) / Hs consumed by previous GEMM2

    // ---------------- GEMM1 chunk: X @ W1t -> gelu -> Hs ----------------
    for (int p = 0; p < 4; ++p) {
      const int ntA = wave * 8 + 2 * p;       // local N-tile in chunk
      const int ntB = ntA + 1;
      const int nlA = ntA * 16 + r16;
      const int nlB = ntB * 16 + r16;
      v8f c00 = {}, c10 = {}, c01 = {}, c11 = {};
      const __bf16* x0 = &Xs[r16 * DIM + kh];
      const __bf16* x1 = &Xs[(16 + r16) * DIM + kh];
      const __bf16* bA = &W1e[(size_t)(hb + nlA) * DIM + kh];
      const __bf16* bB = &W1e[(size_t)(hb + nlB) * DIM + kh];
      for (int k0 = 0; k0 < DIM; k0 += 32) {
        v16bf fa0 = ldfrag(x0 + k0);
        v16bf fa1 = ldfrag(x1 + k0);
        v16bf fbA = ldfrag(bA + k0);
        v16bf fbB = ldfrag(bB + k0);
        c00 = wmma_bf16(fa0, fbA, c00);
        c10 = wmma_bf16(fa1, fbA, c10);
        c01 = wmma_bf16(fa0, fbB, c01);
        c11 = wmma_bf16(fa1, fbB, c11);
      }
      const float biasA = b1[e * HID + hb + nlA];
      const float biasB = b1[e * HID + hb + nlB];
#pragma unroll
      for (int r = 0; r < 8; ++r) {
        int m0 = kh + r, m1 = 16 + kh + r;
        Hs[m0 * HCH + nlA] = (__bf16)gelu_erf(c00[r] + biasA);
        Hs[m1 * HCH + nlA] = (__bf16)gelu_erf(c10[r] + biasA);
        Hs[m0 * HCH + nlB] = (__bf16)gelu_erf(c01[r] + biasB);
        Hs[m1 * HCH + nlB] = (__bf16)gelu_erf(c11[r] + biasB);
      }
    }
    __syncthreads();

    // ---------------- GEMM2 chunk: Ys += Hs @ W2t ----------------
    const bool last = (hb + HCH == HID);
    for (int p = 0; p < 4; ++p) {
      const int ntA = wave * 8 + 2 * p;       // N-tile over D
      const int ntB = ntA + 1;
      const int nA  = ntA * 16 + r16;
      const int nB  = ntB * 16 + r16;
      // Ys fragment slots (unit = nt*2 + msub), 8 floats/lane contiguous
      float* yA0 = &Ys[((ntA * 2 + 0) * 32 + lane) * 8];
      float* yA1 = &Ys[((ntA * 2 + 1) * 32 + lane) * 8];
      float* yB0 = &Ys[((ntB * 2 + 0) * 32 + lane) * 8];
      float* yB1 = &Ys[((ntB * 2 + 1) * 32 + lane) * 8];
      v8f c00, c10, c01, c11;
      if (hb == 0) {
        c00 = (v8f){}; c10 = (v8f){}; c01 = (v8f){}; c11 = (v8f){};
      } else {
        c00 = *(const v8f*)yA0; c10 = *(const v8f*)yA1;
        c01 = *(const v8f*)yB0; c11 = *(const v8f*)yB1;
      }
      const __bf16* h0 = &Hs[r16 * HCH + kh];
      const __bf16* h1 = &Hs[(16 + r16) * HCH + kh];
      const __bf16* bA = &W2e[(size_t)nA * HID + hb + kh];
      const __bf16* bB = &W2e[(size_t)nB * HID + hb + kh];
      for (int k0 = 0; k0 < HCH; k0 += 32) {
        v16bf fa0 = ldfrag(h0 + k0);
        v16bf fa1 = ldfrag(h1 + k0);
        v16bf fbA = ldfrag(bA + k0);
        v16bf fbB = ldfrag(bB + k0);
        c00 = wmma_bf16(fa0, fbA, c00);
        c10 = wmma_bf16(fa1, fbA, c10);
        c01 = wmma_bf16(fa0, fbB, c01);
        c11 = wmma_bf16(fa1, fbB, c11);
      }
      if (!last) {
        *(v8f*)yA0 = c00; *(v8f*)yA1 = c10;
        *(v8f*)yB0 = c01; *(v8f*)yB1 = c11;
      } else {
        // fused epilogue: + b2, gating weight, atomic scatter
        const float biasA = b2[e * DIM + nA];
        const float biasB = b2[e * DIM + nB];
#pragma unroll
        for (int r = 0; r < 8; ++r) {
          int m0 = kh + r, m1 = 16 + kh + r;
          atomicAdd(&out[(size_t)tks[m0] * DIM + nA], wts[m0] * (c00[r] + biasA));
          atomicAdd(&out[(size_t)tks[m1] * DIM + nA], wts[m1] * (c10[r] + biasA));
          atomicAdd(&out[(size_t)tks[m0] * DIM + nB], wts[m0] * (c01[r] + biasB));
          atomicAdd(&out[(size_t)tks[m1] * DIM + nB], wts[m1] * (c11[r] + biasB));
        }
      }
    }
  }
}

// ---------------------------------------------------------------------------
// Launch
// ---------------------------------------------------------------------------
extern "C" void kernel_launch(void* const* d_in, const int* in_sizes, int n_in,
                              void* d_out, int out_size, void* d_ws, size_t ws_size,
                              hipStream_t stream) {
  const float* x  = (const float*)d_in[0];
  const float* Wg = (const float*)d_in[1];
  const float* W1 = (const float*)d_in[2];
  const float* b1 = (const float*)d_in[3];
  const float* W2 = (const float*)d_in[4];
  const float* b2 = (const float*)d_in[5];
  float* out = (float*)d_out;
  char*  ws  = (char*)d_ws;

  int*    cnt   = (int*)  (ws + OFF_CNT);
  float*  fsum  = (float*)(ws + OFF_FSUM);
  float*  psum  = (float*)(ws + OFF_PSUM);
  int*    tokL  = (int*)  (ws + OFF_TOK);
  float*  wL    = (float*)(ws + OFF_WGT);
  __bf16* xbf   = (__bf16*)(ws + OFF_XBF);
  __bf16* w1t   = (__bf16*)(ws + OFF_W1BF);
  __bf16* w2t   = (__bf16*)(ws + OFF_W2BF);

  // zero accumulators + output (out is accumulated by atomics)
  hipMemsetAsync(ws, 0, 256, stream);
  hipMemsetAsync(d_out, 0, (size_t)(T_TOK * DIM + 1) * sizeof(float), stream);

  // fp32 -> bf16 staging (+ weight transpose to [n][k])
  f32_to_bf16<<<2048, 256, 0, stream>>>(x, xbf, (size_t)T_TOK * DIM);
  {
    dim3 g1(HID / TT, DIM / TT, NEXP);   // W1 [D][H] -> W1t [H][D]
    convert_transpose<<<g1, 256, 0, stream>>>(W1, w1t, DIM, HID);
    dim3 g2(DIM / TT, HID / TT, NEXP);   // W2 [H][D] -> W2t [D][H]
    convert_transpose<<<g2, 256, 0, stream>>>(W2, w2t, HID, DIM);
  }

  // routing + aux loss
  router_kernel<<<T_TOK / 256, 256, 0, stream>>>(x, Wg, cnt, fsum, psum, tokL, wL);
  finalize_aux<<<1, 1, 0, stream>>>(fsum, psum, out + (size_t)T_TOK * DIM);

  // expert FFN (~256 KB dynamic LDS; WGP supports 320 KB)
  hipFuncSetAttribute((const void*)expert_ffn,
                      hipFuncAttributeMaxDynamicSharedMemorySize, (int)LDS_TOTAL);
  expert_ffn<<<NEXP * MAXTILE, 256, LDS_TOTAL, stream>>>(
      xbf, w1t, w2t, b1, b2, cnt, tokL, wL, out);
}
